// RouterAugmentedAttention_20177756357135
// MI455X (gfx1250) — compile-verified, matrix operand order checked by reading, not run on hardware
//
#include <hip/hip_runtime.h>
#include <hip/hip_bf16.h>
#include <math.h>

#define B_  8
#define S_  1024
#define H_  12
#define D_  64
#define HID 768
#define KTOP 6

typedef __attribute__((ext_vector_type(16))) __bf16 v16bf;
typedef __attribute__((ext_vector_type(8)))  __bf16 v8bf;
typedef __attribute__((ext_vector_type(8)))  float  v8f;

#if defined(__has_builtin)
#  if __has_builtin(__builtin_amdgcn_global_load_async_to_lds_b128) && \
      __has_builtin(__builtin_amdgcn_s_wait_asynccnt)
#    define USE_ASYNC_V 1
#  endif
#endif
#ifndef USE_ASYNC_V
#  define USE_ASYNC_V 0
#endif

#if USE_ASYNC_V
// Builtin signature (from hipcc diagnostic): arg0 = global (AS1) pointer to
// int __attribute__((vector_size(16))), arg1 = LDS (AS3) pointer, then two
// immediate ints (offset, cpol).
typedef int gccv4i __attribute__((vector_size(16)));
typedef __attribute__((address_space(1))) gccv4i* gp128_t;
typedef __attribute__((address_space(3))) gccv4i* lp128_t;
#endif

__device__ __forceinline__ v8f wmma_bf16(v16bf a, v16bf b, v8f c) {
  // (neg_a, A, neg_b, B, c_mod, C, reuse_a, reuse_b)
  return __builtin_amdgcn_wmma_f32_16x16x32_bf16(false, a, false, b, (short)0, c, false, false);
}

// Build a 16-element bf16 fragment from two contiguous 8-element (16B) groups.
// Per the CDNA5 16-bit A/B layout, lane L holds K = {khalf..khalf+7} and
// {khalf+16..khalf+23} with khalf = (L>=16)*8 -> two aligned v8bf loads.
__device__ __forceinline__ v16bf make_frag(const __bf16* lo_p, const __bf16* hi_p) {
  v8bf lo = *(const v8bf*)lo_p;
  v8bf hi = *(const v8bf*)hi_p;
  return __builtin_shufflevector(lo, hi, 0,1,2,3,4,5,6,7,8,9,10,11,12,13,14,15);
}

// ---------------------------------------------------------------------------
// 1) Router: mean over S, logits = mean @ Wr + br, top-6 hard mask (forward
//    value of head_mask is exactly the hard mask).
// ---------------------------------------------------------------------------
__global__ void __launch_bounds__(256) router_kernel(
    const float* __restrict__ hs, const float* __restrict__ Wr,
    const float* __restrict__ br, float* __restrict__ mask) {
  __shared__ float lmean[HID];
  __shared__ float llog[H_];
  const int b = blockIdx.x;
  const int t = threadIdx.x;
  const float* hp = hs + (size_t)b * S_ * HID;
  float a0 = 0.f, a1 = 0.f, a2 = 0.f;
  for (int s = 0; s < S_; ++s) {
    const float* r = hp + (size_t)s * HID;
    a0 += r[t]; a1 += r[t + 256]; a2 += r[t + 512];
  }
  const float inv = 1.f / (float)S_;
  lmean[t] = a0 * inv; lmean[t + 256] = a1 * inv; lmean[t + 512] = a2 * inv;
  __syncthreads();
  if (t < H_) {
    float acc = br[t];
    for (int c = 0; c < HID; ++c) acc += lmean[c] * Wr[c * H_ + t];
    llog[t] = acc;
  }
  __syncthreads();
  if (t == 0) {
    float lg[H_], mk[H_];
    for (int h = 0; h < H_; ++h) { lg[h] = llog[h]; mk[h] = 0.f; }
    for (int it = 0; it < KTOP; ++it) {
      int best = 0; float bv = lg[0];
      for (int h = 1; h < H_; ++h) if (lg[h] > bv) { bv = lg[h]; best = h; }
      mk[best] = 1.f; lg[best] = -3.0e38f;
    }
    for (int h = 0; h < H_; ++h) mask[b * H_ + h] = mk[h];
  }
}

// ---------------------------------------------------------------------------
// 2) QKV projection: Y = X @ W + b, masked, stored bf16.
//    Block tile 128x64 (one head), 8 waves x (16 rows x 64 cols), K-step 32.
//    Q,K stored (b,h,s,d); V stored transposed (b,h,d,s) for the ctx GEMM.
// ---------------------------------------------------------------------------
__global__ void __launch_bounds__(256) qkv_kernel(
    const float* __restrict__ X,
    const float* __restrict__ Wq, const float* __restrict__ bq,
    const float* __restrict__ Wk, const float* __restrict__ bk,
    const float* __restrict__ Wv, const float* __restrict__ bv,
    const float* __restrict__ mask,
    __bf16* __restrict__ qb, __bf16* __restrict__ kb, __bf16* __restrict__ vb) {
  __shared__ __bf16 lA[128 * 32];   // X tile, row-major (m,k)
  __shared__ __bf16 lB[64 * 32];    // W tile, transposed (n,k)
  const int m0 = blockIdx.x * 128;
  const int h  = blockIdx.y;
  const int n0 = h * 64;
  const int z  = blockIdx.z;
  const float* W    = (z == 0) ? Wq : (z == 1) ? Wk : Wv;
  const float* bias = (z == 0) ? bq : (z == 1) ? bk : bv;
  const int t = threadIdx.x;
  const int wave = t >> 5, lane = t & 31;
  const int col   = lane & 15;
  const int khalf = (lane >> 4) << 3;
  const int rbase = khalf;                 // C/D row base: 0 or 8
  const int ar = t >> 1, akh = (t & 1) << 4;

  v8f acc[4] = {};
  for (int k0 = 0; k0 < HID; k0 += 32) {
    __syncthreads();
    { // stage X tile (128x32), f32 -> bf16
      const float* src = X + (size_t)(m0 + ar) * HID + k0 + akh;
      if (k0 + 32 < HID) __builtin_prefetch(src + 32, 0, 1);   // global_prefetch_b8
      #pragma unroll
      for (int i = 0; i < 16; ++i) lA[ar * 32 + akh + i] = (__bf16)src[i];
    }
    #pragma unroll
    for (int i = 0; i < 8; ++i) {          // stage W tile transposed (64x32)
      int e = t * 8 + i; int kk = e >> 6, nn = e & 63;
      lB[nn * 32 + kk] = (__bf16)W[(size_t)(k0 + kk) * HID + n0 + nn];
    }
    __syncthreads();
    v16bf afrag = make_frag(&lA[(wave * 16 + col) * 32 + khalf],
                            &lA[(wave * 16 + col) * 32 + khalf + 16]);
    v16bf bfrag[4];
    #pragma unroll
    for (int c = 0; c < 4; ++c)            // all frag loads first...
      bfrag[c] = make_frag(&lB[(c * 16 + col) * 32 + khalf],
                           &lB[(c * 16 + col) * 32 + khalf + 16]);
    #pragma unroll
    for (int c = 0; c < 4; ++c)            // ...then a dense WMMA burst
      acc[c] = wmma_bf16(afrag, bfrag[c], acc[c]);
  }
  const int bidx = m0 >> 10;               // 128-row tile stays inside one batch
  const float mval = mask[bidx * H_ + h];
  #pragma unroll
  for (int c = 0; c < 4; ++c) {
    const int d = c * 16 + col;
    const float bv_ = bias[n0 + d];
    #pragma unroll
    for (int r = 0; r < 8; ++r) {
      const int row = m0 + wave * 16 + rbase + r;
      const int s = row & (S_ - 1);
      const float val = (acc[c][r] + bv_) * mval;
      if (z == 2)
        vb[((size_t)(bidx * H_ + h) * D_ + d) * S_ + s] = (__bf16)val;
      else {
        __bf16* dst = (z == 0) ? qb : kb;
        dst[((size_t)(bidx * H_ + h) * S_ + s) * D_ + d] = (__bf16)val;
      }
    }
  }
}

// ---------------------------------------------------------------------------
// 3) Flash attention: one wave owns a 16-row Q tile, streams 32 keys/step.
//    V tile staged LDS via async-to-LDS (issued early, waited late) when the
//    toolchain exposes the gfx1250 async builtins; direct b128 frags otherwise.
// ---------------------------------------------------------------------------
__global__ void __launch_bounds__(256) attn_kernel(
    const __bf16* __restrict__ qb, const __bf16* __restrict__ kb,
    const __bf16* __restrict__ vb, __bf16* __restrict__ ctx) {
  __shared__ __bf16 lp[8 * 16 * 32];       // per-wave P-tile slabs
#if USE_ASYNC_V
  __shared__ __bf16 lv[8 * 64 * 32];       // per-wave V-tile slabs (dcol,kk)
#endif
  const int t = threadIdx.x, wave = t >> 5, lane = t & 31;
  const int col   = lane & 15;
  const int khalf = (lane >> 4) << 3;
  const int rbase = khalf;
  const int qtile = blockIdx.x * 8 + wave;
  const int h = blockIdx.y, b = blockIdx.z;
  const size_t bh = (size_t)b * H_ + h;
  const __bf16* qp = qb + bh * S_ * D_ + (size_t)qtile * 16 * D_;
  const __bf16* kp = kb + bh * S_ * D_;
  const __bf16* vp = vb + bh * D_ * S_;    // transposed (d,s)
  __bf16* lw = lp + wave * 16 * 32;
#if USE_ASYNC_V
  __bf16* lvw = lv + wave * 64 * 32;
#endif

  v16bf qa[2];
  #pragma unroll
  for (int dc = 0; dc < 2; ++dc) {
    const __bf16* p = qp + col * D_ + dc * 32 + khalf;
    qa[dc] = make_frag(p, p + 16);
  }
  v8f o[4] = {};
  float mrow[8], lrow[8];
  #pragma unroll
  for (int r = 0; r < 8; ++r) { mrow[r] = -3.0e38f; lrow[r] = 0.f; }

  for (int j = 0; j < S_; j += 32) {
#if USE_ASYNC_V
    // Kick off the V-tile (64x32) DMA into LDS early: 8 x 16B per lane.
    #pragma unroll
    for (int i = 0; i < 8; ++i) {
      const int e = i * 32 + lane;                 // 16B chunk id
      const __bf16* src = vp + (size_t)(e >> 2) * S_ + j + ((e & 3) << 3);
      __bf16* dst = lvw + e * 8;
      __builtin_amdgcn_global_load_async_to_lds_b128(
          (gp128_t)(size_t)src, (lp128_t)(unsigned)(size_t)dst, 0, 0);
    }
#endif
    // ---- scores: load all K fragments, then 4 WMMAs back-to-back ----
    v16bf kf[2][2];
    #pragma unroll
    for (int tt = 0; tt < 2; ++tt) {
      const __bf16* kbase = kp + (size_t)(j + tt * 16 + col) * D_;
      #pragma unroll
      for (int dc = 0; dc < 2; ++dc) {
        const __bf16* p = kbase + dc * 32 + khalf;
        kf[tt][dc] = make_frag(p, p + 16);
      }
    }
    v8f sc[2] = {};
    #pragma unroll
    for (int tt = 0; tt < 2; ++tt)
      #pragma unroll
      for (int dc = 0; dc < 2; ++dc)
        sc[tt] = wmma_bf16(qa[dc], kf[tt][dc], sc[tt]);

    // ---- online softmax (row stats across the 16-lane C-layout groups) ----
    float p0[8], p1[8], corr[8];
    #pragma unroll
    for (int r = 0; r < 8; ++r) {
      float s0 = sc[0][r] * 0.125f, s1 = sc[1][r] * 0.125f;  // 1/sqrt(64)
      float rm = fmaxf(s0, s1);
      rm = fmaxf(rm, __shfl_xor(rm, 1));
      rm = fmaxf(rm, __shfl_xor(rm, 2));
      rm = fmaxf(rm, __shfl_xor(rm, 4));
      rm = fmaxf(rm, __shfl_xor(rm, 8));
      float nm = fmaxf(mrow[r], rm);
      corr[r] = __expf(mrow[r] - nm);
      p0[r] = __expf(s0 - nm);
      p1[r] = __expf(s1 - nm);
      float rs = p0[r] + p1[r];
      rs += __shfl_xor(rs, 1);
      rs += __shfl_xor(rs, 2);
      rs += __shfl_xor(rs, 4);
      rs += __shfl_xor(rs, 8);
      lrow[r] = lrow[r] * corr[r] + rs;
      mrow[r] = nm;
    }
    #pragma unroll
    for (int c = 0; c < 4; ++c)
      #pragma unroll
      for (int r = 0; r < 8; ++r) o[c][r] *= corr[r];

    // ---- C-layout -> A-layout transpose of P through LDS (same wave) ----
    #pragma unroll
    for (int r = 0; r < 8; ++r) {
      lw[(rbase + r) * 32 + col]      = (__bf16)p0[r];
      lw[(rbase + r) * 32 + 16 + col] = (__bf16)p1[r];
    }
    asm volatile("" ::: "memory");   // same-wave DS ops are in-order; just
                                     // stop compiler reordering of the RAW
    v16bf pa = make_frag(&lw[col * 32 + khalf], &lw[col * 32 + khalf + 16]);

    // ---- ctx: all V fragments, then 4 WMMAs back-to-back ----
    v16bf vf[4];
#if USE_ASYNC_V
    __builtin_amdgcn_s_wait_asynccnt(0);
    asm volatile("" ::: "memory");
    #pragma unroll
    for (int c = 0; c < 4; ++c) {
      const __bf16* p = lvw + (c * 16 + col) * 32 + khalf;
      vf[c] = make_frag(p, p + 16);
    }
#else
    #pragma unroll
    for (int c = 0; c < 4; ++c) {
      const __bf16* p = vp + (size_t)(c * 16 + col) * S_ + j + khalf;
      vf[c] = make_frag(p, p + 16);
    }
#endif
    #pragma unroll
    for (int c = 0; c < 4; ++c) o[c] = wmma_bf16(pa, vf[c], o[c]);
  }

  #pragma unroll
  for (int c = 0; c < 4; ++c) {
    const int d = c * 16 + col;
    #pragma unroll
    for (int r = 0; r < 8; ++r) {
      const int srow = qtile * 16 + rbase + r;
      const float val = o[c][r] / lrow[r];
      ctx[((size_t)(b * S_ + srow)) * HID + h * D_ + d] = (__bf16)val;
    }
  }
}

// ---------------------------------------------------------------------------
// 4) Output projection + bias + residual -> x (f32, written into d_out).
// ---------------------------------------------------------------------------
__global__ void __launch_bounds__(256) oproj_kernel(
    const __bf16* __restrict__ ctx, const float* __restrict__ Wo,
    const float* __restrict__ bo, const float* __restrict__ hs,
    float* __restrict__ xout) {
  __shared__ __bf16 lB[64 * 32];
  const int m0 = blockIdx.x * 128;
  const int n0 = blockIdx.y * 64;
  const int t = threadIdx.x, wave = t >> 5, lane = t & 31;
  const int col = lane & 15, khalf = (lane >> 4) << 3, rbase = khalf;
  v8f acc[4] = {};
  for (int k0 = 0; k0 < HID; k0 += 32) {
    __syncthreads();
    #pragma unroll
    for (int i = 0; i < 8; ++i) {
      int e = t * 8 + i; int kk = e >> 6, nn = e & 63;
      lB[nn * 32 + kk] = (__bf16)Wo[(size_t)(k0 + kk) * HID + n0 + nn];
    }
    __syncthreads();
    const __bf16* ap = ctx + (size_t)(m0 + wave * 16 + col) * HID + k0;
    if (k0 + 32 < HID) __builtin_prefetch(ap + 32, 0, 1);
    v16bf afrag = make_frag(ap + khalf, ap + khalf + 16);
    v16bf bfrag[4];
    #pragma unroll
    for (int c = 0; c < 4; ++c)
      bfrag[c] = make_frag(&lB[(c * 16 + col) * 32 + khalf],
                           &lB[(c * 16 + col) * 32 + khalf + 16]);
    #pragma unroll
    for (int c = 0; c < 4; ++c)
      acc[c] = wmma_bf16(afrag, bfrag[c], acc[c]);
  }
  #pragma unroll
  for (int c = 0; c < 4; ++c) {
    const int n = n0 + c * 16 + col;
    const float bias = bo[n];
    #pragma unroll
    for (int r = 0; r < 8; ++r) {
      const size_t row = (size_t)(m0 + wave * 16 + rbase + r);
      xout[row * HID + n] = acc[c][r] + bias + hs[row * HID + n];
    }
  }
}

// ---------------------------------------------------------------------------
// 5) In-place LayerNorm over the last dim (768), one block per row.
// ---------------------------------------------------------------------------
__global__ void __launch_bounds__(256) ln_kernel(
    float* __restrict__ x, const float* __restrict__ g,
    const float* __restrict__ bta) {
  __shared__ float rs[256], rq[256];
  const int t = threadIdx.x;
  const size_t row = blockIdx.x;
  float v[3];
  #pragma unroll
  for (int i = 0; i < 3; ++i) v[i] = x[row * HID + t + i * 256];
  rs[t] = v[0] + v[1] + v[2];
  rq[t] = v[0] * v[0] + v[1] * v[1] + v[2] * v[2];
  __syncthreads();
  for (int off = 128; off > 0; off >>= 1) {
    if (t < off) { rs[t] += rs[t + off]; rq[t] += rq[t + off]; }
    __syncthreads();
  }
  const float mu  = rs[0] * (1.f / HID);
  const float var = rq[0] * (1.f / HID) - mu * mu;
  const float inv = rsqrtf(var + 1e-12f);
  #pragma unroll
  for (int i = 0; i < 3; ++i) {
    const int c = t + i * 256;
    x[row * HID + c] = (v[i] - mu) * inv * g[c] + bta[c];
  }
}

extern "C" void kernel_launch(void* const* d_in, const int* in_sizes, int n_in,
                              void* d_out, int out_size, void* d_ws, size_t ws_size,
                              hipStream_t stream) {
  const float* hs = (const float*)d_in[0];
  const float* Wq = (const float*)d_in[1];
  const float* bq = (const float*)d_in[2];
  const float* Wk = (const float*)d_in[3];
  const float* bk = (const float*)d_in[4];
  const float* Wv = (const float*)d_in[5];
  const float* bv = (const float*)d_in[6];
  const float* Wr = (const float*)d_in[7];
  const float* br = (const float*)d_in[8];
  const float* Wo = (const float*)d_in[9];
  const float* bo = (const float*)d_in[10];
  const float* lg = (const float*)d_in[11];
  const float* lb = (const float*)d_in[12];
  float* out = (float*)d_out;

  char* ws = (char*)d_ws;
  float* mask = (float*)ws;                      // 96 floats (padded to 512B)
  const size_t qkv_elems = (size_t)B_ * H_ * S_ * D_;
  __bf16* qb  = (__bf16*)(ws + 512);
  __bf16* kb  = qb + qkv_elems;
  __bf16* vb  = kb + qkv_elems;
  __bf16* ctx = vb + qkv_elems;                  // (B,S,HID) bf16

  router_kernel<<<dim3(B_), dim3(256), 0, stream>>>(hs, Wr, br, mask);
  qkv_kernel<<<dim3(64, H_, 3), dim3(256), 0, stream>>>(hs, Wq, bq, Wk, bk, Wv, bv,
                                                        mask, qb, kb, vb);
  attn_kernel<<<dim3(8, H_, B_), dim3(256), 0, stream>>>(qb, kb, vb, ctx);
  oproj_kernel<<<dim3(64, H_), dim3(256), 0, stream>>>(ctx, Wo, bo, hs, out);
  ln_kernel<<<dim3(B_ * S_), dim3(256), 0, stream>>>(out, lg, lb);
}
